// AttentionNetPooling_56839597195673
// MI455X (gfx1250) — matrix operand order repaired and checked by reading, not compiled, not run on hardware
//
#include <hip/hip_runtime.h>
#include <hip/hip_bf16.h>
#include <math.h>

typedef __attribute__((ext_vector_type(16))) __bf16 v16bf;
typedef __attribute__((ext_vector_type(8)))  float  v8f;

#define N_NODES   200000
#define IN_DIM    320
#define HID       128
#define POOL      256
#define NUM_G     2000
#define KSTEPS    10            // 320 / 32
#define NTILES    8             // 128 / 16
#define MTILES    (N_NODES / 16)  // 12500, exact

// ---- workspace layout (units: floats from base of d_ws) ----
#define OFF_BPACK    0          // 40960 bf16 = 20480 floats (B fragments, pre-swizzled)
#define OFF_SCORES   20480      // N_NODES floats
#define OFF_BLOCKMAX 220480     // 256
#define OFF_BLOCKSUM 220736     // 256
#define OFF_GMAX     220992     // 1
#define OFF_GSUM     220993     // 1
#define OFF_SUMS     221056     // NUM_G*POOL = 512000
#define OFF_COUNTS   733056     // NUM_G
#define ZERO_COUNT   (NUM_G * POOL + NUM_G)

// ---------------------------------------------------------------------------
// Pack W1 (fp32 [128][320]) into bf16 B-fragments in the exact WMMA lane
// layout: fragment id = (ks*NTILES + nt)*32 + lane, 16 bf16 per fragment.
// B element e of lane:  k = ks*32 + (e<8 ? 8h+e : 16+8h+(e-8)), h = lane>=16,  n = nt*16 + (lane&15)
// ---------------------------------------------------------------------------
__global__ __launch_bounds__(256) void pack_b_kernel(const float* __restrict__ W1,
                                                     __bf16* __restrict__ Bpack) {
    int i = blockIdx.x * 256 + threadIdx.x;               // element index
    if (i >= KSTEPS * NTILES * 32 * 16) return;
    int e    = i & 15;
    int frag = i >> 4;
    int lane = frag & 31;
    int nt   = (frag >> 5) & 7;
    int ks   = frag >> 8;
    int k = ks * 32 + ((e < 8) ? ((lane >> 4) * 8 + e) : (16 + (lane >> 4) * 8 + (e - 8)));
    int n = nt * 16 + (lane & 15);
    Bpack[i] = (__bf16)W1[n * IN_DIM + k];
}

__global__ __launch_bounds__(256) void zero_accum_kernel(float* __restrict__ ws) {
    int i = blockIdx.x * 256 + threadIdx.x;
    if (i < ZERO_COUNT) ws[OFF_SUMS + i] = 0.0f;
}

// ---------------------------------------------------------------------------
// Score kernel: per wave one 16-node tile.  c[nt] (8 x v8f) accumulates the
// full 16x128 hidden tile across K, seeded with b1.  z is streamed exactly
// once: each lane loads 2x(8 consecutive floats) per k-step (disjoint K
// halves between lane and lane+16), converted fp32->bf16 in-register.
// ---------------------------------------------------------------------------
__global__ __launch_bounds__(256) void score_kernel(const float* __restrict__ z,
                                                    const __bf16* __restrict__ Bpack,
                                                    const float* __restrict__ b1,
                                                    const float* __restrict__ W2,
                                                    const float* __restrict__ b2,
                                                    float* __restrict__ scores) {
    __shared__ float sB1[HID];
    __shared__ float sW2[HID];
    const int tid = threadIdx.x;
    if (tid < HID) { sB1[tid] = b1[tid]; sW2[tid] = W2[tid]; }
    __syncthreads();

    const int wave = tid >> 5;
    const int lane = tid & 31;
    const int tile = blockIdx.x * 8 + wave;
    if (tile >= MTILES) return;                // wave-uniform guard, EXEC stays full

    const int ncol_lane = lane & 15;
    const int hiK       = lane >> 4;           // 0: K lo-halves, 1: K hi-halves
    const int row       = tile * 16 + ncol_lane;
    const float* zrow   = z + (size_t)row * IN_DIM;
    const v16bf* bp     = reinterpret_cast<const v16bf*>(Bpack);

    v8f c[NTILES];
    #pragma unroll
    for (int nt = 0; nt < NTILES; ++nt) {
        float bb = sB1[nt * 16 + ncol_lane];   // bias folded into accumulator
        #pragma unroll
        for (int r = 0; r < 8; ++r) c[nt][r] = bb;
    }

    #pragma unroll
    for (int ks = 0; ks < KSTEPS; ++ks) {
        const int ka0 = ks * 32 + (hiK ? 8 : 0);
        const int ka1 = ks * 32 + (hiK ? 24 : 16);
        float4 p0 = *reinterpret_cast<const float4*>(zrow + ka0);
        float4 p1 = *reinterpret_cast<const float4*>(zrow + ka0 + 4);
        float4 p2 = *reinterpret_cast<const float4*>(zrow + ka1);
        float4 p3 = *reinterpret_cast<const float4*>(zrow + ka1 + 4);
        v16bf a;
        a[0]  = (__bf16)p0.x; a[1]  = (__bf16)p0.y; a[2]  = (__bf16)p0.z; a[3]  = (__bf16)p0.w;
        a[4]  = (__bf16)p1.x; a[5]  = (__bf16)p1.y; a[6]  = (__bf16)p1.z; a[7]  = (__bf16)p1.w;
        a[8]  = (__bf16)p2.x; a[9]  = (__bf16)p2.y; a[10] = (__bf16)p2.z; a[11] = (__bf16)p2.w;
        a[12] = (__bf16)p3.x; a[13] = (__bf16)p3.y; a[14] = (__bf16)p3.z; a[15] = (__bf16)p3.w;
        #pragma unroll
        for (int nt = 0; nt < NTILES; ++nt) {
            v16bf bfrag = bp[(ks * NTILES + nt) * 32 + lane];   // 32B contiguous, L2-hot
            c[nt] = __builtin_amdgcn_wmma_f32_16x16x32_bf16(
                        false, a, false, bfrag, (short)0, c[nt], false, false);
        }
    }

    // fuse relu + W2 dot; C layout: lane holds column n = nt*16 + (lane&15),
    // VGPR r holds row r (lanes 0-15) / row r+8 (lanes 16-31)
    float sacc[8];
    #pragma unroll
    for (int r = 0; r < 8; ++r) sacc[r] = 0.0f;
    #pragma unroll
    for (int nt = 0; nt < NTILES; ++nt) {
        float w2v = sW2[nt * 16 + ncol_lane];
        #pragma unroll
        for (int r = 0; r < 8; ++r) {
            float h = c[nt][r];
            sacc[r] += (h > 0.0f ? h : 0.0f) * w2v;
        }
    }
    // reduce across the 16 lanes of each half (same row set per half)
    #pragma unroll
    for (int r = 0; r < 8; ++r) {
        float v = sacc[r];
        v += __shfl_xor(v, 1, 16);
        v += __shfl_xor(v, 2, 16);
        v += __shfl_xor(v, 4, 16);
        v += __shfl_xor(v, 8, 16);
        sacc[r] = v;
    }
    if (ncol_lane == 0) {
        const float b2v = b2[0];
        int base = tile * 16 + (hiK ? 8 : 0);
        #pragma unroll
        for (int r = 0; r < 8; ++r) scores[base + r] = sacc[r] + b2v;
    }
}

// ---------------------------------------------------------------------------
// Global softmax: max pass, then exp/sum pass (scores overwritten with exp)
// ---------------------------------------------------------------------------
__global__ __launch_bounds__(256) void max_partial_kernel(const float* __restrict__ s,
                                                          float* __restrict__ bmax) {
    __shared__ float sm[256];
    float m = -3.402823466e+38f;
    for (int i = blockIdx.x * 256 + threadIdx.x; i < N_NODES; i += gridDim.x * 256)
        m = fmaxf(m, s[i]);
    sm[threadIdx.x] = m; __syncthreads();
    for (int st = 128; st > 0; st >>= 1) {
        if (threadIdx.x < st) sm[threadIdx.x] = fmaxf(sm[threadIdx.x], sm[threadIdx.x + st]);
        __syncthreads();
    }
    if (threadIdx.x == 0) bmax[blockIdx.x] = sm[0];
}

__global__ __launch_bounds__(256) void max_final_kernel(const float* __restrict__ bmax,
                                                        float* __restrict__ gmax) {
    __shared__ float sm[256];
    sm[threadIdx.x] = bmax[threadIdx.x]; __syncthreads();
    for (int st = 128; st > 0; st >>= 1) {
        if (threadIdx.x < st) sm[threadIdx.x] = fmaxf(sm[threadIdx.x], sm[threadIdx.x + st]);
        __syncthreads();
    }
    if (threadIdx.x == 0) gmax[0] = sm[0];
}

__global__ __launch_bounds__(256) void exp_partial_kernel(float* __restrict__ s,
                                                          const float* __restrict__ gmax,
                                                          float* __restrict__ bsum) {
    __shared__ float sm[256];
    const float gm = gmax[0];
    float acc = 0.0f;
    for (int i = blockIdx.x * 256 + threadIdx.x; i < N_NODES; i += gridDim.x * 256) {
        float e = __expf(s[i] - gm);
        s[i] = e;
        acc += e;
    }
    sm[threadIdx.x] = acc; __syncthreads();
    for (int st = 128; st > 0; st >>= 1) {
        if (threadIdx.x < st) sm[threadIdx.x] += sm[threadIdx.x + st];
        __syncthreads();
    }
    if (threadIdx.x == 0) bsum[blockIdx.x] = sm[0];
}

__global__ __launch_bounds__(256) void sum_final_kernel(const float* __restrict__ bsum,
                                                        float* __restrict__ gsum) {
    __shared__ float sm[256];
    sm[threadIdx.x] = bsum[threadIdx.x]; __syncthreads();
    for (int st = 128; st > 0; st >>= 1) {
        if (threadIdx.x < st) sm[threadIdx.x] += sm[threadIdx.x + st];
        __syncthreads();
    }
    if (threadIdx.x == 0) gsum[0] = sm[0];
}

// ---------------------------------------------------------------------------
// Weighted segment sum: thread = column, block = 256-row chunk.  Sorted
// batch_index -> register accumulation, atomicAdd only at segment boundaries.
// ---------------------------------------------------------------------------
__global__ __launch_bounds__(256) void pool_kernel(const float* __restrict__ z,
                                                   const int* __restrict__ bidx,
                                                   const float* __restrict__ wexp,
                                                   const float* __restrict__ gsum,
                                                   float* __restrict__ sums,
                                                   float* __restrict__ counts) {
    const int c  = threadIdx.x;                 // output column 0..255
    const int r0 = blockIdx.x * 256;
    const int r1 = (r0 + 256 < N_NODES) ? r0 + 256 : N_NODES;
    const float inv = 1.0f / gsum[0];

    int g_prev = bidx[r0];
    float acc = 0.0f, cnt = 0.0f;
    for (int i = r0; i < r1; ++i) {
        int g = bidx[i];
        if (g != g_prev) {
            atomicAdd(&sums[(size_t)g_prev * POOL + c], acc);
            if (c == 0) atomicAdd(&counts[g_prev], cnt);
            acc = 0.0f; cnt = 0.0f; g_prev = g;
        }
        float w = wexp[i] * inv;
        acc += w * z[(size_t)i * IN_DIM + c];   // uses z[:, :256]
        cnt += 1.0f;
    }
    atomicAdd(&sums[(size_t)g_prev * POOL + c], acc);
    if (c == 0) atomicAdd(&counts[g_prev], cnt);
}

__global__ __launch_bounds__(256) void finalize_kernel(const float* __restrict__ sums,
                                                       const float* __restrict__ counts,
                                                       float* __restrict__ out) {
    int i = blockIdx.x * 256 + threadIdx.x;
    if (i >= NUM_G * POOL) return;
    float cnt = counts[i / POOL];
    out[i] = sums[i] / fmaxf(cnt, 1.0f);
}

// ---------------------------------------------------------------------------
extern "C" void kernel_launch(void* const* d_in, const int* in_sizes, int n_in,
                              void* d_out, int out_size, void* d_ws, size_t ws_size,
                              hipStream_t stream) {
    const float* z  = (const float*)d_in[0];
    const int*   bi = (const int*)  d_in[1];
    const float* W1 = (const float*)d_in[2];
    const float* b1 = (const float*)d_in[3];
    const float* W2 = (const float*)d_in[4];
    const float* b2 = (const float*)d_in[5];
    float* out = (float*)d_out;
    float* ws  = (float*)d_ws;

    __bf16* Bpack  = (__bf16*)(ws + OFF_BPACK);
    float* scores  = ws + OFF_SCORES;
    float* bmax    = ws + OFF_BLOCKMAX;
    float* bsum    = ws + OFF_BLOCKSUM;
    float* gmax    = ws + OFF_GMAX;
    float* gsum    = ws + OFF_GSUM;
    float* sums    = ws + OFF_SUMS;
    float* counts  = ws + OFF_COUNTS;

    pack_b_kernel<<<(KSTEPS * NTILES * 32 * 16 + 255) / 256, 256, 0, stream>>>(W1, Bpack);
    zero_accum_kernel<<<(ZERO_COUNT + 255) / 256, 256, 0, stream>>>(ws);

    score_kernel<<<(MTILES + 7) / 8, 256, 0, stream>>>(z, Bpack, b1, W2, b2, scores);

    max_partial_kernel<<<256, 256, 0, stream>>>(scores, bmax);
    max_final_kernel<<<1, 256, 0, stream>>>(bmax, gmax);
    exp_partial_kernel<<<256, 256, 0, stream>>>(scores, gmax, bsum);
    sum_final_kernel<<<1, 256, 0, stream>>>(bsum, gsum);

    pool_kernel<<<(N_NODES + 255) / 256, 256, 0, stream>>>(z, bi, scores, gsum, sums, counts);
    finalize_kernel<<<(NUM_G * POOL + 255) / 256, 256, 0, stream>>>(sums, counts, out);
}
